// Decoder_68152541053662
// MI455X (gfx1250) — compile-verified, matrix operand order checked by reading, not run on hardware
//
#include <hip/hip_runtime.h>
#include <hip/hip_bf16.h>

#define N_NODES   12000
#define N_EDGES   384000
#define NE_TOT    (N_EDGES + N_NODES)   // edges + self loops = 396000
#define NEG_SLOPE 0.2f

typedef __attribute__((ext_vector_type(2)))  float    v2f;
typedef __attribute__((ext_vector_type(8)))  float    v8f;
typedef __attribute__((ext_vector_type(16))) _Float16 v16h;

#if __has_builtin(__builtin_amdgcn_wmma_f32_16x16x4_f32)
#define USE_F32_WMMA 1
#else
#define USE_F32_WMMA 0
#endif

// ---------------- dense mini-GEMM: H[N][FOUT] = X[N][FIN] @ W[FIN][FOUT] ----------------
template <int FIN, int FOUT>
__global__ void gemm_kernel(const float* __restrict__ X, const float* __restrict__ W,
                            float* __restrict__ H) {
    __shared__ float sW[FIN * FOUT];
    for (int i = threadIdx.x; i < FIN * FOUT; i += blockDim.x) sW[i] = W[i];
    __syncthreads();
    int t = blockIdx.x * blockDim.x + threadIdx.x;
    if (t >= N_NODES * FOUT) return;
    int n = t / FOUT, f = t % FOUT;
    const float* xr = X + (size_t)n * FIN;
    float acc = 0.0f;
#pragma unroll
    for (int k = 0; k < FIN; ++k) acc = fmaf(xr[k], sW[k * FOUT + f], acc);
    H[t] = acc;
}

// ------------- per-node attention logits + init of segment max/sum accumulators -------------
template <int F>
__global__ void alpha_kernel(const float* __restrict__ H, const float* __restrict__ a_src,
                             const float* __restrict__ a_dst, float* __restrict__ As,
                             float* __restrict__ Ad, float* __restrict__ M,
                             float* __restrict__ S) {
    int n = blockIdx.x * blockDim.x + threadIdx.x;
    if (n >= N_NODES) return;
    const float* hr = H + (size_t)n * F;
    float s = 0.0f, d = 0.0f;
#pragma unroll
    for (int k = 0; k < F; ++k) {
        s = fmaf(hr[k], a_src[k], s);
        d = fmaf(hr[k], a_dst[k], d);
    }
    As[n] = s;
    Ad[n] = d;
    M[n]  = -__builtin_inff();
    S[n]  = 0.0f;
}

__device__ __forceinline__ void edge_sd(const int* __restrict__ ei, int e, int& s, int& d) {
    if (e < N_EDGES) { s = ei[e]; d = ei[N_EDGES + e]; }
    else             { s = d = e - N_EDGES; }       // appended self loops
}

__device__ __forceinline__ void atomicMaxF(float* addr, float v) {
    // IEEE trick: non-negative floats order like ints, negative like reversed uints.
    if (v >= 0.0f) atomicMax((int*)addr, __float_as_int(v));
    else           atomicMin((unsigned int*)addr, __float_as_uint(v));
}

// ---------------- segment softmax over incoming edges, 3 passes ----------------
__global__ void edge_max_kernel(const int* __restrict__ ei, const float* __restrict__ As,
                                const float* __restrict__ Ad, float* __restrict__ eb,
                                float* __restrict__ M) {
    int e = blockIdx.x * blockDim.x + threadIdx.x;
    if (e >= NE_TOT) return;
    int s, d;
    edge_sd(ei, e, s, d);
    float v = As[s] + Ad[d];
    v = (v > 0.0f) ? v : NEG_SLOPE * v;   // leaky_relu
    eb[e] = v;
    atomicMaxF(M + d, v);
}

__global__ void edge_exp_kernel(const int* __restrict__ ei, float* __restrict__ eb,
                                const float* __restrict__ M, float* __restrict__ S) {
    int e = blockIdx.x * blockDim.x + threadIdx.x;
    if (e >= NE_TOT) return;
    int s, d;
    edge_sd(ei, e, s, d);
    float ex = __expf(eb[e] - M[d]);
    eb[e] = ex;
    atomicAdd(S + d, ex);
}

__global__ void edge_norm_kernel(const int* __restrict__ ei, float* __restrict__ eb,
                                 const float* __restrict__ S) {
    int e = blockIdx.x * blockDim.x + threadIdx.x;
    if (e >= NE_TOT) return;
    int s, d;
    edge_sd(ei, e, s, d);
    eb[e] = eb[e] / S[d];
}

// ---------------- weighted scatter aggregation: O[dst] += alpha * H[src] ----------------
template <int F>
__global__ void aggr_kernel(const int* __restrict__ ei, const float* __restrict__ eb,
                            const float* __restrict__ H, float* __restrict__ O) {
    int t = blockIdx.x * blockDim.x + threadIdx.x;
    if (t >= NE_TOT * F) return;
    int e = t / F, f = t % F;
    int s, d;
    edge_sd(ei, e, s, d);
    atomicAdd(O + (size_t)d * F + f, eb[e] * H[(size_t)s * F + f]);
}

template <int F>
__global__ void bias_relu_kernel(float* __restrict__ O, const float* __restrict__ b) {
    int t = blockIdx.x * blockDim.x + threadIdx.x;
    if (t >= N_NODES * F) return;
    float v = O[t] + b[t % F];
    O[t] = (v > 0.0f) ? v : 0.0f;
}

// z (f32, K=40) -> f16 padded to K=64 (used only by the f16 WMMA fallback path)
__global__ void convert_pad_kernel(const float* __restrict__ z, _Float16* __restrict__ zh) {
    int t = blockIdx.x * blockDim.x + threadIdx.x;
    if (t >= N_NODES * 64) return;
    int n = t >> 6, k = t & 63;
    zh[t] = (k < 40) ? (_Float16)z[(size_t)n * 40 + k] : (_Float16)0.0f;
}

#if !USE_F32_WMMA
__device__ __forceinline__ v8f tile_acc_f16(const _Float16* __restrict__ zh, int i0, int j0,
                                            int lane) {
    const _Float16* zha = zh + (size_t)(i0 + (lane & 15)) * 64;
    const _Float16* zhb = zh + (size_t)(j0 + (lane & 15)) * 64;
    const int b0 = (lane >> 4) << 3;
    v8f acc = {};
#pragma unroll
    for (int kb = 0; kb < 2; ++kb) {
        v16h a, b;
#pragma unroll
        for (int j = 0; j < 8; ++j) {
            a[j]     = zha[kb * 32 + b0 + j];
            a[j + 8] = zha[kb * 32 + 16 + b0 + j];
            b[j]     = zhb[kb * 32 + b0 + j];
            b[j + 8] = zhb[kb * 32 + 16 + b0 + j];
        }
        acc = __builtin_amdgcn_wmma_f32_16x16x32_f16(false, a, false, b, (short)0, acc,
                                                     false, false);
    }
    return acc;
}
#endif

// ---------------- decoder: out = sigmoid(z @ z^T), one 16x32 tile per wave ----------------
__global__ __launch_bounds__(160) void decoder_wmma_kernel(const float* __restrict__ z,
                                                           const _Float16* __restrict__ zh,
                                                           float* __restrict__ out) {
    const int lane = threadIdx.x;                         // wave32: lane id
    const int ti = blockIdx.y;                            // 0..749  (16-row tile)
    const int tj = blockIdx.x * blockDim.y + threadIdx.y; // 0..374  (32-col tile), exact fit
    const int i0 = ti << 4, j0 = tj << 5;
    const int r  = lane & 15;
    v8f acc0 = {}, acc1 = {};
#if USE_F32_WMMA
    // f32 A 16x4 layout: lane L holds (M=L&15, K = kb*4 + 2*(L>>4) + {0,1})  -> one b64 load.
    // A fragment shared across both column tiles; B fragments mirror A's pattern for z@z^T.
    const int koff = (lane >> 4) << 1;
    const float* za  = z + (size_t)(i0 + r) * 40 + koff;
    const float* zb0 = z + (size_t)(j0 + r) * 40 + koff;
    const float* zb1 = zb0 + 16 * 40;
#pragma unroll
    for (int kb = 0; kb < 10; ++kb) {                     // K = 40 = 10 * 4
        v2f a  = *(const v2f*)(za + kb * 4);
        v2f b0 = *(const v2f*)(zb0 + kb * 4);
        v2f b1 = *(const v2f*)(zb1 + kb * 4);
        acc0 = __builtin_amdgcn_wmma_f32_16x16x4_f32(false, a, false, b0, (short)0, acc0,
                                                     false, false);
        acc1 = __builtin_amdgcn_wmma_f32_16x16x4_f32(false, a, false, b1, (short)0, acc1,
                                                     false, false);
    }
#else
    acc0 = tile_acc_f16(zh, i0, j0, lane);
    acc1 = tile_acc_f16(zh, i0, j0 + 16, lane);
#endif
    // D layout: VGPR v -> row i0 + v + 8*(lane>>4); lane&15 -> column.
    const int row0 = i0 + ((lane >> 4) << 3);
    const int col  = j0 + r;
#pragma unroll
    for (int v = 0; v < 8; ++v) {
        size_t rowbase = (size_t)(row0 + v) * N_NODES;
        float s0 = __builtin_amdgcn_rcpf(1.0f + __expf(-acc0[v]));  // fast sigmoid
        float s1 = __builtin_amdgcn_rcpf(1.0f + __expf(-acc1[v]));
        __builtin_nontemporal_store(s0, out + rowbase + col);        // streaming (NT) stores
        __builtin_nontemporal_store(s1, out + rowbase + col + 16);
    }
}

extern "C" void kernel_launch(void* const* d_in, const int* in_sizes, int n_in,
                              void* d_out, int out_size, void* d_ws, size_t ws_size,
                              hipStream_t stream) {
    const float* x    = (const float*)d_in[0];
    const int*   ei   = (const int*)d_in[1];   // edge_index [2, E]
    const float* W1   = (const float*)d_in[2];
    const float* a_s1 = (const float*)d_in[3];
    const float* a_d1 = (const float*)d_in[4];
    const float* b1   = (const float*)d_in[5];
    const float* W2   = (const float*)d_in[6];
    const float* a_s2 = (const float*)d_in[7];
    const float* a_d2 = (const float*)d_in[8];
    const float* b2   = (const float*)d_in[9];
    float* out = (float*)d_out;

    // workspace carve-up (float units, 256-elem aligned) ~= 12 MB total
    float* ws  = (float*)d_ws;
    size_t off = 0;
    auto alloc = [&](size_t nf) {
        float* p = ws + off;
        off += (nf + 255) & ~(size_t)255;
        return p;
    };
    float* h1 = alloc((size_t)N_NODES * 50);
    float* As = alloc(N_NODES);
    float* Ad = alloc(N_NODES);
    float* M  = alloc(N_NODES);
    float* S  = alloc(N_NODES);
    float* eb = alloc(NE_TOT);
    float* o1 = alloc((size_t)N_NODES * 50);
    float* h2 = alloc((size_t)N_NODES * 40);
    float* o2 = alloc((size_t)N_NODES * 40);               // z
    _Float16* zh = (_Float16*)alloc((size_t)N_NODES * 32); // 12000 x 64 halves

    const int B = 256;
    auto cdiv = [](long long a, long long b) { return (unsigned)((a + b - 1) / b); };

    // ---------------- GAT layer 1: 40 -> 50 ----------------
    gemm_kernel<40, 50><<<cdiv((long long)N_NODES * 50, B), B, 0, stream>>>(x, W1, h1);
    alpha_kernel<50><<<cdiv(N_NODES, B), B, 0, stream>>>(h1, a_s1, a_d1, As, Ad, M, S);
    edge_max_kernel<<<cdiv(NE_TOT, B), B, 0, stream>>>(ei, As, Ad, eb, M);
    edge_exp_kernel<<<cdiv(NE_TOT, B), B, 0, stream>>>(ei, eb, M, S);
    edge_norm_kernel<<<cdiv(NE_TOT, B), B, 0, stream>>>(ei, eb, S);
    hipMemsetAsync(o1, 0, (size_t)N_NODES * 50 * sizeof(float), stream);
    aggr_kernel<50><<<cdiv((long long)NE_TOT * 50, B), B, 0, stream>>>(ei, eb, h1, o1);
    bias_relu_kernel<50><<<cdiv((long long)N_NODES * 50, B), B, 0, stream>>>(o1, b1);

    // ---------------- GAT layer 2: 50 -> 40 ----------------
    gemm_kernel<50, 40><<<cdiv((long long)N_NODES * 40, B), B, 0, stream>>>(o1, W2, h2);
    alpha_kernel<40><<<cdiv(N_NODES, B), B, 0, stream>>>(h2, a_s2, a_d2, As, Ad, M, S);
    edge_max_kernel<<<cdiv(NE_TOT, B), B, 0, stream>>>(ei, As, Ad, eb, M);
    edge_exp_kernel<<<cdiv(NE_TOT, B), B, 0, stream>>>(ei, eb, M, S);
    edge_norm_kernel<<<cdiv(NE_TOT, B), B, 0, stream>>>(ei, eb, S);
    hipMemsetAsync(o2, 0, (size_t)N_NODES * 40 * sizeof(float), stream);
    aggr_kernel<40><<<cdiv((long long)NE_TOT * 40, B), B, 0, stream>>>(ei, eb, h2, o2);
    bias_relu_kernel<40><<<cdiv((long long)N_NODES * 40, B), B, 0, stream>>>(o2, b2);

    // ---------------- inner-product decoder via WMMA ----------------
    convert_pad_kernel<<<cdiv((long long)N_NODES * 64, B), B, 0, stream>>>(o2, zh);
    dim3 dgrid(75, 750);    // 75 * 5 = 375 column tiles (32 wide), 750 row tiles (16 tall)
    dim3 dblk(32, 5);       // 5 waves per block, one 16x32 tile per wave
    decoder_wmma_kernel<<<dgrid, dblk, 0, stream>>>(o2, zh, out);
}
